// GIN2_40132174414143
// MI455X (gfx1250) — matrix operand order repaired and checked by reading, not compiled, not run on hardware
//
#include <hip/hip_runtime.h>

// ---------------- CDNA5 WMMA types ----------------
typedef __attribute__((ext_vector_type(16))) __bf16 v16bf;
typedef __attribute__((ext_vector_type(8)))  float  v8f;

#define N_NODES 100000
#define N_EDGES 1600000
#define N_PAIRS 100000
#define DIM_IN  128
#define DIM_H   128
#define DIM_MLP 512
#define DIM_PRED 64

#if defined(__has_builtin)
#if __has_builtin(__builtin_amdgcn_cvt_pk_bf16_f32)
#define HAVE_HW_PKBF16 1
#endif
#endif

// f32 -> bf16, round-to-nearest-even (software fallback)
__device__ __forceinline__ unsigned short f2bf(float f) {
    unsigned u = __float_as_uint(f);
    u += 0x7FFFu + ((u >> 16) & 1u);
    return (unsigned short)(u >> 16);
}

// pack two f32 -> two bf16 in one dword (hardware cvt when available)
__device__ __forceinline__ unsigned pk2(float a, float b) {
#ifdef HAVE_HW_PKBF16
    auto r = __builtin_amdgcn_cvt_pk_bf16_f32(a, b);   // a -> [15:0], b -> [31:16]
    unsigned u;
    __builtin_memcpy(&u, &r, 4);
    return u;
#else
    return (unsigned)f2bf(a) | ((unsigned)f2bf(b) << 16);
#endif
}

struct __align__(16) U4 { unsigned int x, y, z, w; };

union AV {
    v16bf v;
    unsigned short u[16];
    unsigned int   w[8];
    U4 q[2];
};

__device__ __forceinline__ v8f wmma_bf16(v16bf a, v16bf b, v8f c) {
    // D = A(16x32 bf16) * B(32x16 bf16) + C(16x16 f32)
    return __builtin_amdgcn_wmma_f32_16x16x32_bf16(false, a, false, b,
                                                   (short)0, c, false, false);
}

// ---------------- helper kernels ----------------

// agg = h  (vectorized copy; fuses GIN self term, eps=0)
__global__ void copy4_kernel(const float4* __restrict__ in,
                             float4* __restrict__ out, int n4) {
    int i = blockIdx.x * blockDim.x + threadIdx.x;
    if (i < n4) out[i] = in[i];
}

// agg[dst] += h[src]  — 32 threads/edge, each handles a float4 of features.
// h and agg are L2-resident (51 MB each << 192 MB L2), so these atomics hit L2.
__global__ void scatter_kernel(const float* __restrict__ h,
                               const int* __restrict__ src,
                               const int* __restrict__ dst,
                               float* __restrict__ agg, int E) {
    int idx = blockIdx.x * blockDim.x + threadIdx.x;
    if (idx >= E * 32) return;
    int e = idx >> 5;
    int q = idx & 31;
    int s = src[e];
    int d = dst[e];
    float4 v = *(const float4*)(h + (size_t)s * DIM_IN + q * 4);
    float* p = agg + (size_t)d * DIM_IN + q * 4;
    atomicAdd(p + 0, v.x);
    atomicAdd(p + 1, v.y);
    atomicAdd(p + 2, v.z);
    atomicAdd(p + 3, v.w);
}

// Pre-swizzle an f32 [K][Nc] weight into the WMMA B-layout as packed bf16:
//   out[ ((kt*ntn + nt)*32 + lane)*16 + i ] = bf16( W[kt*32 + (lane>>4)*8 + (i&7) + (i>>3)*16][nt*16 + (lane&15)] )
// so each lane fetches its whole B fragment as two contiguous 16B loads.
__global__ void pack_b_kernel(const float* __restrict__ W,
                              unsigned short* __restrict__ out, int K, int Nc) {
    int idx = blockIdx.x * blockDim.x + threadIdx.x;
    if (idx >= K * Nc) return;
    int i    = idx & 15;
    int lane = (idx >> 4) & 31;
    int t    = idx >> 9;
    int ntn  = Nc >> 4;
    int kt   = t / ntn;
    int nt   = t - kt * ntn;
    int lrow = lane & 15, lhalf = lane >> 4;
    int k = kt * 32 + lhalf * 8 + (i & 7) + (i >> 3) * 16;
    int n = nt * 16 + lrow;
    out[idx] = f2bf(W[(size_t)k * Nc + n]);
}

// ---------------- fused GIN MLP: h_out = relu( relu(agg@W1 + b1) @ W2 + b2 ) ----------------
// One wave = one 16-node tile. 2 waves/block. z staged as bf16 in LDS (16KB/wave).
__global__ void __launch_bounds__(64) gin_mlp_kernel(
    const float* __restrict__ agg,
    const unsigned short* __restrict__ W1p, const float* __restrict__ b1,
    const unsigned short* __restrict__ W2p, const float* __restrict__ b2,
    float* __restrict__ hout, int ntiles)
{
    __shared__ unsigned short zsh[2][16 * DIM_MLP];   // 32 KB total
    const int wave = threadIdx.x >> 5;
    const int lane = threadIdx.x & 31;
    const int tile = blockIdx.x * 2 + wave;
    if (tile >= ntiles) return;
    const int lrow  = lane & 15;
    const int lhalf = lane >> 4;
    const int node0 = tile * 16;

    // ---- A (16x128) in WMMA A-layout, bf16, kept in registers ----
    v16bf areg[4];
    const float* arow = agg + (size_t)(node0 + lrow) * DIM_IN;
#pragma unroll
    for (int kt = 0; kt < 4; ++kt) {
        AV a;
        const float* p = arow + kt * 32 + lhalf * 8;
#pragma unroll
        for (int i = 0; i < 4; ++i) a.w[i]     = pk2(p[2 * i],      p[2 * i + 1]);
#pragma unroll
        for (int i = 0; i < 4; ++i) a.w[4 + i] = pk2(p[16 + 2 * i], p[16 + 2 * i + 1]);
        areg[kt] = a.v;
    }

    unsigned short* zb = &zsh[wave][0];

    // ---- GEMM1: z = relu(A @ W1 + b1)  (16 x 512) ----
    // Two n-tiles per iteration -> two independent WMMA accumulation chains.
#pragma unroll 1
    for (int nt = 0; nt < 32; nt += 2) {
        float bias0 = b1[nt * 16 + lrow];
        float bias1 = b1[nt * 16 + 16 + lrow];
        v8f acc0, acc1;
#pragma unroll
        for (int j = 0; j < 8; ++j) { acc0[j] = bias0; acc1[j] = bias1; }
#pragma unroll
        for (int kt = 0; kt < 4; ++kt) {
            const v16bf* bp0 =
                (const v16bf*)(W1p + (size_t)((kt * 32 + nt) * 32 + lane) * 16);
            const v16bf* bp1 =
                (const v16bf*)(W1p + (size_t)((kt * 32 + nt + 1) * 32 + lane) * 16);
            acc0 = wmma_bf16(areg[kt], *bp0, acc0);
            acc1 = wmma_bf16(areg[kt], *bp1, acc1);
        }
#pragma unroll
        for (int j = 0; j < 8; ++j) {
            float v0 = fmaxf(acc0[j], 0.0f);             // ReLU
            float v1 = fmaxf(acc1[j], 0.0f);
            int row = (j + lhalf * 8) * DIM_MLP + lrow;
            zb[row + nt * 16]      = (unsigned short)pk2(v0, v0);
            zb[row + nt * 16 + 16] = (unsigned short)pk2(v1, v1);
        }
    }

    // ---- GEMM2: h = relu(z @ W2 + b2)  (16 x 128) ----
    v8f acc2[8];
#pragma unroll
    for (int nt = 0; nt < 8; ++nt) {
        float bias = b2[nt * 16 + lrow];
#pragma unroll
        for (int j = 0; j < 8; ++j) acc2[nt][j] = bias;
    }
#pragma unroll 1
    for (int kt = 0; kt < 16; ++kt) {
        AV a;  // A fragment of z from LDS: two 16B ds loads per lane
        const unsigned short* zp = zb + lrow * DIM_MLP + kt * 32 + lhalf * 8;
        a.q[0] = *(const U4*)(zp);
        a.q[1] = *(const U4*)(zp + 16);
#pragma unroll
        for (int nt = 0; nt < 8; ++nt) {
            const v16bf* bp =
                (const v16bf*)(W2p + (size_t)((kt * 8 + nt) * 32 + lane) * 16);
            acc2[nt] = wmma_bf16(a.v, *bp, acc2[nt]);
        }
    }
#pragma unroll
    for (int nt = 0; nt < 8; ++nt) {
#pragma unroll
        for (int j = 0; j < 8; ++j) {
            int m = j + lhalf * 8;
            hout[(size_t)(node0 + m) * DIM_H + nt * 16 + lrow] =
                fmaxf(acc2[nt][j], 0.0f);                // outer ReLU
        }
    }
}

// ---------------- predictor: out = relu((h[s]*h[d]) @ Wp1 + bp1) @ Wp2 + bp2 ----------------
__global__ void __launch_bounds__(64) gin_pred_kernel(
    const float* __restrict__ h,
    const int* __restrict__ srcI, const int* __restrict__ dstI,
    const unsigned short* __restrict__ Wp1p, const float* __restrict__ bp1,
    const float* __restrict__ Wp2, const float* __restrict__ bp2,
    float* __restrict__ out, int ptiles)
{
    __shared__ float zsh[2][16 * DIM_PRED];   // 8 KB
    const int wave = threadIdx.x >> 5;
    const int lane = threadIdx.x & 31;
    const int tile = blockIdx.x * 2 + wave;
    if (tile >= ptiles) return;
    const int lrow  = lane & 15;
    const int lhalf = lane >> 4;

    const int e = tile * 16 + lrow;
    const int s = srcI[e];
    const int d = dstI[e];
    const float* hs = h + (size_t)s * DIM_H;
    const float* hd = h + (size_t)d * DIM_H;

    v16bf areg[4];   // A = bf16( h[s] * h[d] ), 16 edges x 128 feats
#pragma unroll
    for (int kt = 0; kt < 4; ++kt) {
        AV a;
        int kb = kt * 32 + lhalf * 8;
#pragma unroll
        for (int i = 0; i < 4; ++i)
            a.w[i] = pk2(hs[kb + 2 * i] * hd[kb + 2 * i],
                         hs[kb + 2 * i + 1] * hd[kb + 2 * i + 1]);
#pragma unroll
        for (int i = 0; i < 4; ++i)
            a.w[4 + i] = pk2(hs[kb + 16 + 2 * i] * hd[kb + 16 + 2 * i],
                             hs[kb + 16 + 2 * i + 1] * hd[kb + 16 + 2 * i + 1]);
        areg[kt] = a.v;
    }

    float* zb = &zsh[wave][0];
    // 4 n-tiles: two independent chains per pass
#pragma unroll
    for (int nt = 0; nt < 4; nt += 2) {
        float bias0 = bp1[nt * 16 + lrow];
        float bias1 = bp1[nt * 16 + 16 + lrow];
        v8f acc0, acc1;
#pragma unroll
        for (int j = 0; j < 8; ++j) { acc0[j] = bias0; acc1[j] = bias1; }
#pragma unroll
        for (int kt = 0; kt < 4; ++kt) {
            const v16bf* bp0 =
                (const v16bf*)(Wp1p + (size_t)((kt * 4 + nt) * 32 + lane) * 16);
            const v16bf* bp1v =
                (const v16bf*)(Wp1p + (size_t)((kt * 4 + nt + 1) * 32 + lane) * 16);
            acc0 = wmma_bf16(areg[kt], *bp0, acc0);
            acc1 = wmma_bf16(areg[kt], *bp1v, acc1);
        }
#pragma unroll
        for (int j = 0; j < 8; ++j) {
            int row = (j + lhalf * 8) * DIM_PRED + lrow;
            zb[row + nt * 16]      = fmaxf(acc0[j], 0.0f);
            zb[row + nt * 16 + 16] = fmaxf(acc1[j], 0.0f);
        }
    }

    // final [64 x 1] GEMM: per-edge dot; same-wave LDS ordering guarantees visibility
    if (lane < 16) {
        float sum = bp2[0];
#pragma unroll
        for (int k = 0; k < DIM_PRED; ++k)
            sum += zb[lrow * DIM_PRED + k] * Wp2[k];
        out[tile * 16 + lrow] = sum;
    }
}

// ---------------- launch ----------------
extern "C" void kernel_launch(void* const* d_in, const int* in_sizes, int n_in,
                              void* d_out, int out_size, void* d_ws, size_t ws_size,
                              hipStream_t stream)
{
    const float* x   = (const float*)d_in[0];
    const float* W1  = (const float*)d_in[1];
    const float* b1  = (const float*)d_in[2];
    const float* W2  = (const float*)d_in[3];
    const float* b2  = (const float*)d_in[4];
    const float* Wp1 = (const float*)d_in[5];
    const float* bp1 = (const float*)d_in[6];
    const float* Wp2 = (const float*)d_in[7];
    const float* bp2 = (const float*)d_in[8];
    const int* esrc  = (const int*)d_in[9];
    const int* edst  = (const int*)d_in[10];
    const int* psrc  = (const int*)d_in[11];
    const int* pdst  = (const int*)d_in[12];
    const int* nsrc  = (const int*)d_in[13];
    const int* ndst  = (const int*)d_in[14];

    float* outp = (float*)d_out;
    float* hpos = outp;
    float* hneg = outp + N_PAIRS;
    float* hfin = outp + 2 * N_PAIRS;    // final h lives directly in d_out

    // workspace layout
    float* agg = (float*)d_ws;                                   // N*128 f32
    float* hA  = agg + (size_t)N_NODES * DIM_H;                  // N*128 f32
    unsigned short* W1p  = (unsigned short*)(hA + (size_t)N_NODES * DIM_H);
    unsigned short* W2p  = W1p + (size_t)3 * DIM_IN * DIM_MLP;
    unsigned short* Wp1p = W2p + (size_t)3 * DIM_MLP * DIM_H;

    // pre-swizzle weights into WMMA B-layout (bf16)
    for (int l = 0; l < 3; ++l) {
        pack_b_kernel<<<(DIM_IN * DIM_MLP + 255) / 256, 256, 0, stream>>>(
            W1 + (size_t)l * DIM_IN * DIM_MLP, W1p + (size_t)l * DIM_IN * DIM_MLP,
            DIM_IN, DIM_MLP);
        pack_b_kernel<<<(DIM_MLP * DIM_H + 255) / 256, 256, 0, stream>>>(
            W2 + (size_t)l * DIM_MLP * DIM_H, W2p + (size_t)l * DIM_MLP * DIM_H,
            DIM_MLP, DIM_H);
    }
    pack_b_kernel<<<(DIM_H * DIM_PRED + 255) / 256, 256, 0, stream>>>(
        Wp1, Wp1p, DIM_H, DIM_PRED);

    const int n4     = N_NODES * DIM_H / 4;
    const int ntiles = N_NODES / 16;     // 6250
    const int ptiles = N_PAIRS / 16;     // 6250

    for (int l = 0; l < 3; ++l) {
        const float* hin = (l == 0) ? x : hA;
        float* hout = (l == 2) ? hfin : hA;   // MLP reads only agg, so hA->hA is safe
        copy4_kernel<<<(n4 + 255) / 256, 256, 0, stream>>>(
            (const float4*)hin, (float4*)agg, n4);
        scatter_kernel<<<(N_EDGES * 32 + 255) / 256, 256, 0, stream>>>(
            hin, esrc, edst, agg, N_EDGES);
        gin_mlp_kernel<<<(ntiles + 1) / 2, 64, 0, stream>>>(
            agg,
            W1p + (size_t)l * DIM_IN * DIM_MLP, b1 + l * DIM_MLP,
            W2p + (size_t)l * DIM_MLP * DIM_H,  b2 + l * DIM_H,
            hout, ntiles);
    }

    gin_pred_kernel<<<(ptiles + 1) / 2, 64, 0, stream>>>(
        hfin, psrc, pdst, Wp1p, bp1, Wp2, bp2, hpos, ptiles);
    gin_pred_kernel<<<(ptiles + 1) / 2, 64, 0, stream>>>(
        hfin, nsrc, ndst, Wp1p, bp1, Wp2, bp2, hneg, ptiles);
}
